// FCN_DFConv_87703232184536
// MI455X (gfx1250) — matrix operand order repaired and checked by reading, not compiled, not run on hardware
//
#include <hip/hip_runtime.h>
#include <stddef.h>

// Problem constants (from reference setup_inputs)
#define Hd   96
#define Wd   96
#define HWd  (Hd * Wd)        // 9216
#define Bd   2
#define Cd   256
#define NPOS (Bd * HWd)       // 18432
#define KKd  9
#define KSTEPS 72             // 9 kk * 8 channel-chunks of 32

typedef __attribute__((ext_vector_type(16))) _Float16 v16h;
typedef __attribute__((ext_vector_type(8)))  _Float16 v8h;
typedef __attribute__((ext_vector_type(8)))  float    v8f;

// ---------------------------------------------------------------------------
// fp32 weights [Osrc, 256, 3, 3] -> f16 WMMA A-fragment layout:
// dst flat = (((kk*8 + cc)*MiTot + mi)*32 + lane)*16 + slot
// A 16x32 f16 fragment: lane<16 -> m=lane,   K slots {0..7,16..23}
//                       lane>=16-> m=lane-16,K slots {8..15,24..31}
// ---------------------------------------------------------------------------
__global__ void convert_w_kernel(const float* __restrict__ src,
                                 _Float16* __restrict__ dst,
                                 int Osrc, int MiTot) {
    int i = blockIdx.x * blockDim.x + threadIdx.x;
    int total = 9 * 8 * MiTot * 512;
    if (i >= total) return;
    int slot  = i & 15;
    int lane  = (i >> 4) & 31;
    int rest  = i >> 9;
    int mi    = rest % MiTot;
    int rest2 = rest / MiTot;
    int cc    = rest2 & 7;
    int kk    = rest2 >> 3;
    int half  = lane >> 4;
    int k     = (slot < 8) ? (slot + half * 8) : (slot + 8 + half * 8);
    int o     = mi * 16 + (lane & 15);
    int c     = cc * 32 + k;
    float v   = (o < Osrc) ? src[(o * Cd + c) * 9 + kk] : 0.0f;
    dst[i] = (_Float16)v;
}

// ---------------------------------------------------------------------------
// Tiled NCHW -> NHWC transpose: src [B, C, HW] -> dst [B, HW, C]
// ---------------------------------------------------------------------------
__global__ __launch_bounds__(256)
void nchw_to_nhwc_kernel(const float* __restrict__ src, float* __restrict__ dst) {
    __shared__ float tile[32][33];
    const int tx = threadIdx.x & 31;
    const int ty = threadIdx.x >> 5;
    const int hw0 = blockIdx.x * 32;
    const int c0  = blockIdx.y * 32;
    const int b   = blockIdx.z;
    const float* s = src + (size_t)b * Cd * HWd;
#pragma unroll
    for (int j = 0; j < 4; ++j) {
        int cl = ty + j * 8;
        tile[cl][tx] = s[(size_t)(c0 + cl) * HWd + hw0 + tx];
    }
    __syncthreads();
    float* d = dst + (size_t)b * HWd * Cd;
#pragma unroll
    for (int j = 0; j < 4; ++j) {
        int hwl = ty + j * 8;
        d[(size_t)(hw0 + hwl) * Cd + c0 + tx] = tile[tx][hwl];
    }
}

// ---------------------------------------------------------------------------
// Sample descriptors. Corner indices are pre-multiplied by C (NHWC stride).
// layout: sd[(b*9 + kk)*HW + hw]
// ---------------------------------------------------------------------------
__global__ void sd_identity_kernel(int4* __restrict__ sdI, float4* __restrict__ sdW) {
    int i = blockIdx.x * blockDim.x + threadIdx.x;
    if (i >= Bd * KKd * HWd) return;
    int hw = i % HWd;
    int t  = i / HWd;
    int kk = t % KKd;
    int h  = hw / Wd, w = hw % Wd;
    int y  = h + kk / 3 - 1;
    int x  = w + kk % 3 - 1;
    bool valid = ((unsigned)y < (unsigned)Hd) && ((unsigned)x < (unsigned)Wd);
    sdI[i] = make_int4(valid ? (y * Wd + x) * Cd : 0, 0, 0, 0);
    sdW[i] = make_float4(valid ? 1.0f : 0.0f, 0.0f, 0.0f, 0.0f);
}

__global__ void sd_deform_kernel(const float* __restrict__ om,
                                 int4* __restrict__ sdI, float4* __restrict__ sdW) {
    int i = blockIdx.x * blockDim.x + threadIdx.x;
    if (i >= Bd * KKd * HWd) return;
    int hw = i % HWd;
    int t  = i / HWd;
    int kk = t % KKd;
    int b  = t / KKd;
    int h  = hw / Wd, w = hw % Wd;

    const float* omb = om + (size_t)b * 27 * HWd;
    float dy = omb[(2 * kk + 0) * HWd + hw];
    float dx = omb[(2 * kk + 1) * HWd + hw];
    float mk = omb[(18 + kk) * HWd + hw];
    mk = 1.0f / (1.0f + __expf(-mk));

    float ys = (float)(h + kk / 3 - 1) + dy;
    float xs = (float)(w + kk % 3 - 1) + dx;
    float y0f = floorf(ys), x0f = floorf(xs);
    int y0 = (int)y0f, x0 = (int)x0f;
    int y1 = y0 + 1, x1 = x0 + 1;
    float wy1 = ys - y0f, wx1 = xs - x0f;
    float wy0 = 1.0f - wy1, wx0 = 1.0f - wx1;

    bool vy0 = (unsigned)y0 < (unsigned)Hd, vy1 = (unsigned)y1 < (unsigned)Hd;
    bool vx0 = (unsigned)x0 < (unsigned)Wd, vx1 = (unsigned)x1 < (unsigned)Wd;
    int cy0 = min(max(y0, 0), Hd - 1), cy1 = min(max(y1, 0), Hd - 1);
    int cx0 = min(max(x0, 0), Wd - 1), cx1 = min(max(x1, 0), Wd - 1);

    sdI[i] = make_int4((cy0 * Wd + cx0) * Cd, (cy0 * Wd + cx1) * Cd,
                       (cy1 * Wd + cx0) * Cd, (cy1 * Wd + cx1) * Cd);
    sdW[i] = make_float4((vy0 && vx0) ? wy0 * wx0 * mk : 0.0f,
                         (vy0 && vx1) ? wy0 * wx1 * mk : 0.0f,
                         (vy1 && vx0) ? wy1 * wx0 * mk : 0.0f,
                         (vy1 && vx1) ? wy1 * wx1 * mk : 0.0f);
}

// ---------------------------------------------------------------------------
// Fused deformable implicit GEMM via WMMA, NHWC activations.
// Tile: (MW*64) oc x 64 pos per workgroup, 8 wave32.
// Each wave: MW M-subtiles x 2 N-subtiles -> 2*MW wmma per K-step.
// MW=4 => one workgroup covers all 256 output channels, so every bilinear
// corner is gathered exactly once across the whole GEMM.
// Double-buffered LDS B staging: 1 barrier / K-step.
// ---------------------------------------------------------------------------
template <int MW, bool IDENT>
__global__ __launch_bounds__(256)
void dgemm_deform_kernel(const float* __restrict__ XT,      // [B, HW, C]
                         const _Float16* __restrict__ WA,
                         const int4* __restrict__ sdI,
                         const float4* __restrict__ sdW,
                         float* __restrict__ OUT, int OCw,
                         const float* __restrict__ bias, int relu) {
    __shared__ int4   s_sdi[576];                       // [kk][p], idx * C
    __shared__ float4 s_sdw[576];
    __shared__ int    s_pb[64];                         // b * HW * C
    __shared__ __align__(32) _Float16 ldsB[2][4 * 32 * 16];  // [buf][ns][lane][slot]

    const int tid   = threadIdx.x;
    const int lane  = tid & 31;
    const int wave  = tid >> 5;
    const int MiTot = gridDim.x * 4 * MW;
    const int miBase = (blockIdx.x * 4 + (wave & 3)) * MW;
    const int ns0   = (wave >> 2) * 2;
    const int posBase = blockIdx.y * 64;

    for (int e = tid; e < 576; e += 256) {
        int p  = e & 63;
        int kk = e >> 6;
        int pos = posBase + p;
        int b  = pos / HWd;
        int hw = pos - b * HWd;
        int g  = (b * KKd + kk) * HWd + hw;
        s_sdi[e] = sdI[g];
        s_sdw[e] = sdW[g];
    }
    if (tid < 64) {
        int pos = posBase + tid;
        s_pb[tid] = (pos / HWd) * (HWd * Cd);
    }
    __syncthreads();

    const int q = tid >> 6;   // channel octet 0..3 within 32-chunk
    const int p = tid & 63;   // tile position
    const int laneB = (p & 15) + ((q & 1) ? 16 : 0);
    const int slotB = (q >> 1) * 8;
    _Float16* ldsDst0 = &ldsB[0][(((p >> 4) * 32 + laneB) << 4) + slotB];
    _Float16* ldsDst1 = &ldsB[1][(((p >> 4) * 32 + laneB) << 4) + slotB];
    const float* xbase = XT + s_pb[p] + q * 8;          // hoisted gather base

    // Stage one 32ch x 64pos B-tile chunk (f16) for K-step s into buffer buf.
    auto stage = [&](int s, int buf) {
        int kk = s >> 3;
        int cc = s & 7;
        int4   id = s_sdi[kk * 64 + p];
        float4 wv = s_sdw[kk * 64 + p];
        const float* ptr = xbase + cc * 32;
        float4 aL = *(const float4*)(ptr + id.x);
        float4 aH = *(const float4*)(ptr + id.x + 4);
        float v0, v1, v2, v3, v4, v5, v6, v7;
        if (IDENT) {
            v0 = wv.x * aL.x; v1 = wv.x * aL.y; v2 = wv.x * aL.z; v3 = wv.x * aL.w;
            v4 = wv.x * aH.x; v5 = wv.x * aH.y; v6 = wv.x * aH.z; v7 = wv.x * aH.w;
        } else {
            float4 bL = *(const float4*)(ptr + id.y);
            float4 bH = *(const float4*)(ptr + id.y + 4);
            float4 cL = *(const float4*)(ptr + id.z);
            float4 cH = *(const float4*)(ptr + id.z + 4);
            float4 dL = *(const float4*)(ptr + id.w);
            float4 dH = *(const float4*)(ptr + id.w + 4);
            v0 = wv.x * aL.x + wv.y * bL.x + wv.z * cL.x + wv.w * dL.x;
            v1 = wv.x * aL.y + wv.y * bL.y + wv.z * cL.y + wv.w * dL.y;
            v2 = wv.x * aL.z + wv.y * bL.z + wv.z * cL.z + wv.w * dL.z;
            v3 = wv.x * aL.w + wv.y * bL.w + wv.z * cL.w + wv.w * dL.w;
            v4 = wv.x * aH.x + wv.y * bH.x + wv.z * cH.x + wv.w * dH.x;
            v5 = wv.x * aH.y + wv.y * bH.y + wv.z * cH.y + wv.w * dH.y;
            v6 = wv.x * aH.z + wv.y * bH.z + wv.z * cH.z + wv.w * dH.z;
            v7 = wv.x * aH.w + wv.y * bH.w + wv.z * cH.w + wv.w * dH.w;
        }
        v8h pk;
        pk[0] = (_Float16)v0; pk[1] = (_Float16)v1;
        pk[2] = (_Float16)v2; pk[3] = (_Float16)v3;
        pk[4] = (_Float16)v4; pk[5] = (_Float16)v5;
        pk[6] = (_Float16)v6; pk[7] = (_Float16)v7;
        *(v8h*)(buf ? ldsDst1 : ldsDst0) = pk;
    };

    v8f acc[MW][2];
#pragma unroll
    for (int u = 0; u < MW; ++u) { acc[u][0] = (v8f){}; acc[u][1] = (v8f){}; }

    stage(0, 0);
    __syncthreads();

    for (int s = 0; s < KSTEPS; ++s) {
        if (s + 1 < KSTEPS) stage(s + 1, (s + 1) & 1);

        int buf = s & 1;
        v16h a[MW];
#pragma unroll
        for (int u = 0; u < MW; ++u)
            a[u] = *(const v16h*)(WA +
                    ((size_t)((s * MiTot + miBase + u) * 32 + lane) << 4));
        v16h b0 = *(const v16h*)&ldsB[buf][((ns0 * 32 + lane) << 4)];
        v16h b1 = *(const v16h*)&ldsB[buf][(((ns0 + 1) * 32 + lane) << 4)];
#pragma unroll
        for (int u = 0; u < MW; ++u) {
            acc[u][0] = __builtin_amdgcn_wmma_f32_16x16x32_f16(
                false, a[u], false, b0, (short)0, acc[u][0], false, false);
            acc[u][1] = __builtin_amdgcn_wmma_f32_16x16x32_f16(
                false, a[u], false, b1, (short)0, acc[u][1], false, false);
        }
        __syncthreads();
    }

    // epilogue: C-fragment -> NCHW fp32 (+bias, +ReLU)
    const int n     = lane & 15;
    const int mbase = (lane >> 4) * 8;
#pragma unroll
    for (int u = 0; u < MW; ++u) {
#pragma unroll
        for (int f = 0; f < 2; ++f) {
            int pos = posBase + (ns0 + f) * 16 + n;
            int b   = pos / HWd;
            int hw  = pos - b * HWd;
#pragma unroll
            for (int r = 0; r < 8; ++r) {
                int oc = (miBase + u) * 16 + mbase + r;
                if (oc < OCw) {
                    float v = acc[u][f][r];
                    if (bias) v += bias[oc];
                    if (relu) v = fmaxf(v, 0.0f);
                    OUT[((size_t)b * OCw + oc) * HWd + hw] = v;
                }
            }
        }
    }
}

// ---------------------------------------------------------------------------
extern "C" void kernel_launch(void* const* d_in, const int* in_sizes, int n_in,
                              void* d_out, int out_size, void* d_ws, size_t ws_size,
                              hipStream_t stream) {
    const float* x      = (const float*)d_in[0];
    const float* w_off0 = (const float*)d_in[1];
    const float* b_off0 = (const float*)d_in[2];
    const float* w0     = (const float*)d_in[3];
    const float* w_off1 = (const float*)d_in[4];
    const float* b_off1 = (const float*)d_in[5];
    const float* w1     = (const float*)d_in[6];
    float* out = (float*)d_out;

    char* ws = (char*)d_ws;
    size_t off = 0;
    auto carve = [&](size_t bytes) -> void* {
        void* pp = ws + off;
        off = (off + bytes + 255) & ~(size_t)255;
        return pp;
    };
    const int NSD = Bd * KKd * HWd;                       // 165888
    int4*      sdI_id = (int4*)     carve((size_t)NSD * 16);
    float4*    sdW_id = (float4*)   carve((size_t)NSD * 16);
    int4*      sdI_df = (int4*)     carve((size_t)NSD * 16);
    float4*    sdW_df = (float4*)   carve((size_t)NSD * 16);
    _Float16*  waOff  = (_Float16*) carve((size_t)64  * 2304 * 2);
    _Float16*  waMain = (_Float16*) carve((size_t)256 * 2304 * 2);
    float*     om     = (float*)    carve((size_t)Bd * 27 * HWd * 4);
    float*     xT     = (float*)    carve((size_t)Bd * HWd * Cd * 4);
    float*     hbuf   = (float*)    carve((size_t)Bd * Cd * HWd * 4);
    float*     hT     = (float*)    carve((size_t)Bd * HWd * Cd * 4);
    (void)ws_size; (void)in_sizes; (void)n_in; (void)out_size;

    const int TPB = 256;
    const int gSD = (NSD + TPB - 1) / TPB;
    const int gWoff  = (9 * 8 * 4  * 512 + TPB - 1) / TPB;   // MiTot=4  (Opad=64)
    const int gWmain = (9 * 8 * 16 * 512 + TPB - 1) / TPB;   // MiTot=16 (Opad=256)
    const dim3 gTr(HWd / 32, Cd / 32, Bd);
    const dim3 gemmOff(1, NPOS / 64);    // MW=1: 64-padded 27 out channels
    const dim3 gemmMain(1, NPOS / 64);   // MW=4: TM=256, full M in one WG

    sd_identity_kernel<<<gSD, TPB, 0, stream>>>(sdI_id, sdW_id);
    nchw_to_nhwc_kernel<<<gTr, TPB, 0, stream>>>(x, xT);

    // ---------------- Layer 1 ----------------
    convert_w_kernel<<<gWoff,  TPB, 0, stream>>>(w_off0, waOff, 27, 4);
    convert_w_kernel<<<gWmain, TPB, 0, stream>>>(w0, waMain, 256, 16);
    dgemm_deform_kernel<1, true><<<gemmOff, TPB, 0, stream>>>(
        xT, waOff, sdI_id, sdW_id, om, 27, b_off0, 0);
    sd_deform_kernel<<<gSD, TPB, 0, stream>>>(om, sdI_df, sdW_df);
    dgemm_deform_kernel<4, false><<<gemmMain, TPB, 0, stream>>>(
        xT, waMain, sdI_df, sdW_df, hbuf, 256, nullptr, 1);
    nchw_to_nhwc_kernel<<<gTr, TPB, 0, stream>>>(hbuf, hT);

    // ---------------- Layer 2 ----------------
    convert_w_kernel<<<gWoff,  TPB, 0, stream>>>(w_off1, waOff, 27, 4);
    convert_w_kernel<<<gWmain, TPB, 0, stream>>>(w1, waMain, 256, 16);
    dgemm_deform_kernel<1, true><<<gemmOff, TPB, 0, stream>>>(
        hT, waOff, sdI_id, sdW_id, om, 27, b_off1, 0);
    sd_deform_kernel<<<gSD, TPB, 0, stream>>>(om, sdI_df, sdW_df);
    dgemm_deform_kernel<4, false><<<gemmMain, TPB, 0, stream>>>(
        hT, waMain, sdI_df, sdW_df, out, 256, nullptr, 1);
}
